// FSPT_PatchCore_42056319762925
// MI455X (gfx1250) — matrix operand order repaired and checked by reading, not compiled, not run on hardware
//
#include <hip/hip_runtime.h>
#include <hip/hip_bf16.h>
#include <math.h>

// CDNA5 / gfx1250 wave32 WMMA types
typedef __attribute__((ext_vector_type(16))) __bf16 v16bf;
typedef __attribute__((ext_vector_type(8)))  float  v8f;

#define MTILES  2                   // 16-row M tiles per wave (B reuse factor)
#define QT      (16 * MTILES)       // queries per block = 32
#define NBLK    (3136 / QT)         // 98 blocks
#define CDIM    768                 // feature dim (K)
#define MROWS   65536               // memory bank rows
#define NCHUNKS (MROWS / 16)        // 4096 column chunks of 16
#define WAVES   8                   // 256 threads / wave32
#define ROWP    776                 // padded bf16 LDS row stride (1552B, bank-rotating)
#define AHALF   (16 * ROWP * 2)     // bytes per precision level per tile = 24832
#define AELEM   (16 * ROWP)         // bf16 elements per precision level per tile
// LDS layout: [tile0-hi][tile1-hi][tile0-lo][tile1-lo]
// -> all ds offsets from a hi-base / lo-base stay < 27KB (16-bit imm field)

union Frag  { uint4 q[2]; v16bf v; };
union Pack4 { __bf16 b[4]; uint2 u; };

// ---------------------------------------------------------------------------
// Prep pass: fp32 bank -> bf16 hi + bf16 lo + fp32 row norms (one wave / row)
// ---------------------------------------------------------------------------
__global__ __launch_bounds__(256)
void bank_prep_kernel(const float* __restrict__ bank,
                      __bf16* __restrict__ hiOut,
                      __bf16* __restrict__ loOut,
                      float*  __restrict__ y2Out)
{
    const int row  = blockIdx.x * WAVES + (threadIdx.x >> 5);
    const int lane = threadIdx.x & 31;

    const float* src = bank  + (size_t)row * CDIM;
    __bf16*      dh  = hiOut + (size_t)row * CDIM;
    __bf16*      dl  = loOut + (size_t)row * CDIM;

    float ssq = 0.0f;
    #pragma unroll
    for (int i = 0; i < CDIM / 128; ++i) {           // 6 iters, 4 floats/lane
        int k = lane * 4 + i * 128;
        float4 f = *(const float4*)(src + k);
        Pack4 ph, pl;
        float fs[4] = { f.x, f.y, f.z, f.w };
        #pragma unroll
        for (int j = 0; j < 4; ++j) {
            ssq += fs[j] * fs[j];
            __bf16 h = (__bf16)fs[j];
            ph.b[j] = h;
            pl.b[j] = (__bf16)(fs[j] - (float)h);
        }
        *(uint2*)(dh + k) = ph.u;
        *(uint2*)(dl + k) = pl.u;
    }
    #pragma unroll
    for (int s = 16; s >= 1; s >>= 1) ssq += __shfl_xor(ssq, s, 32);
    if (lane == 0) y2Out[row] = ssq;
}

// ---------------------------------------------------------------------------
// Main pass: each wave runs 2 M-tiles against shared B fragments.
// PRE = true : bank pre-converted (zero conversion VALU in hot loop)
// PRE = false: convert fp32 bank on the fly (fallback if d_ws too small)
// ---------------------------------------------------------------------------
template<bool PRE>
__global__ __launch_bounds__(256)
void patchcore_topk_kernel(const float*  __restrict__ patch,
                           const float*  __restrict__ bank,
                           const __bf16* __restrict__ bankHi,
                           const __bf16* __restrict__ bankLo,
                           const float*  __restrict__ bankY2,
                           float*        __restrict__ out)
{
    // hi tiles then lo tiles (99328 B); reused after the main loop as merge
    // scratch (32 rows * 640 candidates * 4B = 81920 B).
    __shared__ __align__(16) unsigned char smem[2 * MTILES * AHALF];
    __shared__ float x2s[QT];

    const int tid  = threadIdx.x;
    const int tile = blockIdx.x;

    for (int i = tid; i < QT; i += 256) x2s[i] = 0.0f;
    __syncthreads();

    // stage the 32x768 query block into LDS as bf16 hi/lo; fp32 row norms
    for (int i = tid; i < QT * CDIM; i += 256) {
        int row = i / CDIM;
        int k   = i - row * CDIM;
        int tt  = row >> 4;
        int r   = row & 15;
        float f = patch[(size_t)(tile * QT + row) * CDIM + k];
        __bf16 h = (__bf16)f;
        __bf16 l = (__bf16)(f - (float)h);
        ((__bf16*)smem)[tt * AELEM + r * ROWP + k]                   = h;
        ((__bf16*)smem)[MTILES * AELEM + tt * AELEM + r * ROWP + k]  = l;
        atomicAdd(&x2s[row], f * f);          // ds_add_f32
    }
    __syncthreads();

    const int wave  = tid >> 5;
    const int lane  = tid & 31;
    const int nsel  = lane & 15;              // B column / A row selector
    const int hi16  = lane >> 4;              // wave32 lane half
    const int abase = hi16 * 8;               // row-half for C (rows 8*hi16..+7)

    // two lane base pointers; every fragment is an immediate ds offset from one
    const __bf16* aHiBase = (const __bf16*)smem + nsel * ROWP + hi16 * 16;
    const __bf16* aLoBase = aHiBase + MTILES * AELEM;

    float x2r[MTILES][8];
    #pragma unroll
    for (int tt = 0; tt < MTILES; ++tt)
        #pragma unroll
        for (int r = 0; r < 8; ++r) x2r[tt][r] = x2s[tt * 16 + abase + r];

    float t[MTILES][8][5];                    // per-lane sorted top-5 per C row
    #pragma unroll
    for (int tt = 0; tt < MTILES; ++tt)
        #pragma unroll
        for (int r = 0; r < 8; ++r)
            #pragma unroll
            for (int j = 0; j < 5; ++j) t[tt][r][j] = 3.0e38f;

    for (int chunk = wave; chunk < NCHUNKS; chunk += WAVES) {
        const size_t rowoff = (size_t)((chunk << 4) + nsel) * CDIM;
        v8f   acc[MTILES] = {};
        float ssq = 0.0f;
        float y2v = 0.0f;

        if constexpr (PRE) {
            const __bf16* bh = bankHi + rowoff + hi16 * 16;
            const __bf16* bl = bankLo + rowoff + hi16 * 16;
            if (chunk + WAVES < NCHUNKS) {
                __builtin_prefetch(bankHi + rowoff + (size_t)WAVES * 16 * CDIM, 0, 1);
                __builtin_prefetch(bankLo + rowoff + (size_t)WAVES * 16 * CDIM, 0, 1);
            }
            y2v = bankY2[(chunk << 4) + nsel];

            #pragma unroll 2
            for (int ki = 0; ki < CDIM / 32; ++ki) {
                // shared B fragments: 16 consecutive bf16 (32B) per level
                const uint4* pbh = (const uint4*)(bh + ki * 32);
                const uint4* pbl = (const uint4*)(bl + ki * 32);
                Frag bhf, blf;
                bhf.q[0] = pbh[0]; bhf.q[1] = pbh[1];
                blf.q[0] = pbl[0]; blf.q[1] = pbl[1];

                #pragma unroll
                for (int tt = 0; tt < MTILES; ++tt) {
                    const uint4* ph = (const uint4*)(aHiBase + tt * AELEM + ki * 32);
                    const uint4* pl = (const uint4*)(aLoBase + tt * AELEM + ki * 32);
                    Frag ah, al;
                    ah.q[0] = ph[0]; ah.q[1] = ph[1];   // K = hi16*16+0..15
                    al.q[0] = pl[0]; al.q[1] = pl[1];

                    acc[tt] = __builtin_amdgcn_wmma_f32_16x16x32_bf16(
                        false, ah.v, false, bhf.v, (short)0, acc[tt], false, false);
                    acc[tt] = __builtin_amdgcn_wmma_f32_16x16x32_bf16(
                        false, ah.v, false, blf.v, (short)0, acc[tt], false, false);
                    acc[tt] = __builtin_amdgcn_wmma_f32_16x16x32_bf16(
                        false, al.v, false, bhf.v, (short)0, acc[tt], false, false);
                }
            }
        } else {
            const float* bptr = bank + rowoff + hi16 * 16;
            #pragma unroll 2
            for (int ki = 0; ki < CDIM / 32; ++ki) {
                const float4* p = (const float4*)(bptr + ki * 32);
                float4 f0 = p[0], f1 = p[1], f2 = p[2], f3 = p[3];
                float fs[16] = { f0.x, f0.y, f0.z, f0.w,  f1.x, f1.y, f1.z, f1.w,
                                 f2.x, f2.y, f2.z, f2.w,  f3.x, f3.y, f3.z, f3.w };
                v16bf bhi, blo;
                #pragma unroll
                for (int j = 0; j < 16; ++j) {
                    ssq += fs[j] * fs[j];
                    __bf16 h = (__bf16)fs[j];
                    bhi[j] = h;
                    blo[j] = (__bf16)(fs[j] - (float)h);
                }
                #pragma unroll
                for (int tt = 0; tt < MTILES; ++tt) {
                    const uint4* ph = (const uint4*)(aHiBase + tt * AELEM + ki * 32);
                    const uint4* pl = (const uint4*)(aLoBase + tt * AELEM + ki * 32);
                    Frag ah, al;
                    ah.q[0] = ph[0]; ah.q[1] = ph[1];
                    al.q[0] = pl[0]; al.q[1] = pl[1];

                    acc[tt] = __builtin_amdgcn_wmma_f32_16x16x32_bf16(
                        false, ah.v, false, bhi, (short)0, acc[tt], false, false);
                    acc[tt] = __builtin_amdgcn_wmma_f32_16x16x32_bf16(
                        false, ah.v, false, blo, (short)0, acc[tt], false, false);
                    acc[tt] = __builtin_amdgcn_wmma_f32_16x16x32_bf16(
                        false, al.v, false, bhi, (short)0, acc[tt], false, false);
                }
            }
            y2v = ssq + __shfl_xor(ssq, 16, 32); // partner lane: other K-half
        }

        // distances + branchless sorted top-5 insert
        #pragma unroll
        for (int tt = 0; tt < MTILES; ++tt)
            #pragma unroll
            for (int r = 0; r < 8; ++r) {
                float d2 = x2r[tt][r] + y2v - 2.0f * acc[tt][r];
                float v  = sqrtf(fmaxf(d2, 0.0f));
                #pragma unroll
                for (int j = 0; j < 5; ++j) {
                    float lo = fminf(t[tt][r][j], v);
                    float hi = fmaxf(t[tt][r][j], v);
                    t[tt][r][j] = lo;
                    v = hi;
                }
            }
    }

    // ---- merge: A LDS region is dead now; reuse it for candidates ----
    __syncthreads();
    float* mg = (float*)smem;
    #pragma unroll
    for (int tt = 0; tt < MTILES; ++tt)
        #pragma unroll
        for (int r = 0; r < 8; ++r) {
            int row = tt * 16 + abase + r;
            #pragma unroll
            for (int j = 0; j < 5; ++j)
                mg[row * 640 + wave * 80 + nsel * 5 + j] = t[tt][r][j];
        }
    __syncthreads();

    if (tid < QT) {
        float best[5];
        #pragma unroll
        for (int j = 0; j < 5; ++j) best[j] = 3.0e38f;
        const float* rp = mg + tid * 640;
        for (int i = 0; i < 640; ++i) {
            float v = rp[i];
            #pragma unroll
            for (int j = 0; j < 5; ++j) {
                float lo = fminf(best[j], v);
                float hi = fmaxf(best[j], v);
                best[j] = lo;
                v = hi;
            }
        }
        float d1  = best[0];
        float gap = fmaxf(best[4] - d1, 0.0f);
        out[tile * QT + tid] = d1 * (1.0f - expf(-gap));
    }
}

extern "C" void kernel_launch(void* const* d_in, const int* in_sizes, int n_in,
                              void* d_out, int out_size, void* d_ws, size_t ws_size,
                              hipStream_t stream)
{
    const float* patch = (const float*)d_in[0];   // (16,196,768) fp32
    const float* bank  = (const float*)d_in[1];   // (65536,768)  fp32
    float* out = (float*)d_out;                   // 3136 fp32

    const size_t elems  = (size_t)MROWS * CDIM;
    const size_t needWs = elems * 2 * sizeof(__bf16) + MROWS * sizeof(float);

    if (ws_size >= needWs) {
        __bf16* hi = (__bf16*)d_ws;
        __bf16* lo = hi + elems;
        float*  y2 = (float*)(lo + elems);
        bank_prep_kernel<<<dim3(MROWS / WAVES), dim3(256), 0, stream>>>(bank, hi, lo, y2);
        patchcore_topk_kernel<true><<<dim3(NBLK), dim3(256), 0, stream>>>(
            patch, bank, hi, lo, y2, out);
    } else {
        patchcore_topk_kernel<false><<<dim3(NBLK), dim3(256), 0, stream>>>(
            patch, bank, nullptr, nullptr, nullptr, out);
    }
}
